// Quantizer1d_46943992545688
// MI455X (gfx1250) — matrix operand order repaired and checked by reading, not compiled, not run on hardware
//
#include <hip/hip_runtime.h>

// ---------------------------------------------------------------------------
// VQ (nearest-codebook) for MI455X / gfx1250, wave32.
// Score matmul runs on V_WMMA_F32_16X16X32_BF16 with a bf16 hi/lo split
// (3-term: xh*wh + xh*wl + xl*wh) giving ~fp32-accurate scores at 8x the
// K-per-issue of the f32 WMMA. Gather/losses use the exact f32 tensors.
//
//   x:      [4,4096,512] f32  (16384 tokens)
//   weight: [1024,512]   f32  (codebook)
// Outputs (flat f32, reference return order):
//   quant_out [16384*512] | codebook_loss [1] | commitment_loss [1] | idx [16384]
//
// argmin(x^2 - 2xW + w^2) == argmax(xW - 0.5*w^2); quant_out == gather(W, idx).
// ---------------------------------------------------------------------------

typedef __attribute__((ext_vector_type(4)))  __bf16 v4bf;
typedef __attribute__((ext_vector_type(8)))  __bf16 v8bf;
typedef __attribute__((ext_vector_type(16))) __bf16 v16bf;
typedef __attribute__((ext_vector_type(8)))  float  v8f;

#define DIM      512
#define K_CODES  1024
#define M_TOTAL  16384            // 4 * 4096 tokens
#define QUANT_N  (M_TOTAL * DIM)  // 8388608

// ---------------------------------------------------------------------------
// Kernel 1: per code row (one wave each):
//   nh2[k]  = -0.5 * sum_c w[k][c]^2                  (exact f32)
//   wh/wl   = bf16 hi/lo split of the row (row-major) for WMMA A-fragments
// ---------------------------------------------------------------------------
__global__ __launch_bounds__(256) void vq_prep(const float* __restrict__ w,
                                               float* __restrict__ nh2,
                                               __bf16* __restrict__ wh,
                                               __bf16* __restrict__ wl) {
  const int lane = threadIdx.x & 31;
  const int code = blockIdx.x * 8 + (threadIdx.x >> 5);
  const float*  row = w  + (size_t)code * DIM;
  __bf16* hrow = wh + (size_t)code * DIM;
  __bf16* lrow = wl + (size_t)code * DIM;
  float s = 0.f;
  #pragma unroll
  for (int i = 0; i < 4; ++i) {
    const int c4 = i * 32 + lane;             // 128 float4 per row
    float4 v = ((const float4*)row)[c4];
    s += v.x * v.x + v.y * v.y + v.z * v.z + v.w * v.w;
    v4bf h, l;
    h[0] = (__bf16)v.x; l[0] = (__bf16)(v.x - (float)h[0]);
    h[1] = (__bf16)v.y; l[1] = (__bf16)(v.y - (float)h[1]);
    h[2] = (__bf16)v.z; l[2] = (__bf16)(v.z - (float)h[2]);
    h[3] = (__bf16)v.w; l[3] = (__bf16)(v.w - (float)h[3]);
    *(v4bf*)(hrow + c4 * 4) = h;
    *(v4bf*)(lrow + c4 * 4) = l;
  }
  #pragma unroll
  for (int off = 16; off > 0; off >>= 1) s += __shfl_xor(s, off);
  if (lane == 0) nh2[code] = -0.5f * s;
}

// ---------------------------------------------------------------------------
// Kernel 2: one block per 16-token tile. The block stages its x tile into LDS
// as bf16 hi/lo; wave g scores code group g (128 codes) with 3-term bf16
// WMMAs; argmax in-lane -> shfl_xor(16) -> LDS across waves; then 256 threads
// cooperatively gather the winning f32 rows and accumulate the loss.
// ---------------------------------------------------------------------------
__global__ __launch_bounds__(256) void vq_main(const float* __restrict__ x,
                                               const float* __restrict__ w,
                                               const float* __restrict__ nh2,
                                               const __bf16* __restrict__ wh,
                                               const __bf16* __restrict__ wl,
                                               float* __restrict__ out,
                                               float* __restrict__ partial) {
  __shared__ __align__(32) __bf16 xh[16][DIM];   // 16 KB
  __shared__ __align__(32) __bf16 xl[16][DIM];   // 16 KB
  __shared__ float sbest[8][16];
  __shared__ int   sidx[8][16];
  __shared__ int   fidx[16];
  __shared__ float ssum[8];

  const int tid    = threadIdx.x;
  const int lane   = tid & 31;
  const int g      = tid >> 5;      // wave id == code group (128 codes each)
  const int nrow   = lane & 15;     // lane row within half
  const int h2     = lane >> 4;     // lane half (selects K sub-range)
  const int m_base = blockIdx.x * 16;
  const int cbase  = g * 128;

  // ---- Stage x tile to LDS as bf16 hi/lo (2048 float4, 8 per thread) ----
  #pragma unroll
  for (int it = 0; it < 8; ++it) {
    const int e  = it * 256 + tid;
    const int m  = e >> 7;          // token row 0..15
    const int c4 = e & 127;         // float4 column
    float4 v = ((const float4*)(x + (size_t)(m_base + m) * DIM))[c4];
    v4bf h, l;
    h[0] = (__bf16)v.x; l[0] = (__bf16)(v.x - (float)h[0]);
    h[1] = (__bf16)v.y; l[1] = (__bf16)(v.y - (float)h[1]);
    h[2] = (__bf16)v.z; l[2] = (__bf16)(v.z - (float)h[2]);
    h[3] = (__bf16)v.w; l[3] = (__bf16)(v.w - (float)h[3]);
    *(v4bf*)&xh[m][c4 * 4] = h;
    *(v4bf*)&xl[m][c4 * 4] = l;
  }
  __syncthreads();

  // ---- Accumulators seeded with -0.5*||w||^2 (C layout: VGPR r + lane half
  //      selects code = cbase + t*16 + h2*8 + r, broadcast across the half).
  v8f acc[8];
  #pragma unroll
  for (int t = 0; t < 8; ++t) {
    const int code0 = cbase + t * 16 + h2 * 8;
    #pragma unroll
    for (int r = 0; r < 8; ++r) acc[t][r] = nh2[code0 + r];
  }

  // A operand rows (codes on M): lane nrow -> code cbase + t*16 + nrow.
  const __bf16* whbase = wh + (size_t)(cbase + nrow) * DIM;
  const __bf16* wlbase = wl + (size_t)(cbase + nrow) * DIM;
  // B operand rows (tokens on N): lane nrow -> LDS token row nrow.
  const __bf16* xhrow = &xh[nrow][0];
  const __bf16* xlrow = &xl[nrow][0];

  // ---- K loop: 16 chunks of 32. B frags from LDS feed 8 tiles x 3 WMMAs.
  for (int k0 = 0; k0 < DIM; k0 += 32) {
    // 16-bit B layout: elems 0..7 = K(k0+8*h2)..+7, elems 8..15 = +16 more (2x ds_b128)
    v16bf bh = __builtin_shufflevector(*(const v8bf*)(xhrow + k0 + 8 * h2), *(const v8bf*)(xhrow + k0 + 16 + 8 * h2), 0, 1, 2, 3, 4, 5, 6, 7, 8, 9, 10, 11, 12, 13, 14, 15);
    v16bf bl = __builtin_shufflevector(*(const v8bf*)(xlrow + k0 + 8 * h2), *(const v8bf*)(xlrow + k0 + 16 + 8 * h2), 0, 1, 2, 3, 4, 5, 6, 7, 8, 9, 10, 11, 12, 13, 14, 15);
    #pragma unroll
    for (int t = 0; t < 8; ++t) {
      const __bf16* hr = whbase + (size_t)t * 16 * DIM;
      const __bf16* lr = wlbase + (size_t)t * 16 * DIM;
      // 16-bit A layout: elems 0..7 = K(k0+8*h2)..+7, elems 8..15 = +16 more.
      v8bf a0 = *(const v8bf*)(hr + k0 + 8 * h2);
      v8bf a1 = *(const v8bf*)(hr + k0 + 16 + 8 * h2);
      v16bf ah = __builtin_shufflevector(a0, a1, 0, 1, 2, 3, 4, 5, 6, 7,
                                         8, 9, 10, 11, 12, 13, 14, 15);
      v8bf b0 = *(const v8bf*)(lr + k0 + 8 * h2);
      v8bf b1 = *(const v8bf*)(lr + k0 + 16 + 8 * h2);
      v16bf al = __builtin_shufflevector(b0, b1, 0, 1, 2, 3, 4, 5, 6, 7,
                                         8, 9, 10, 11, 12, 13, 14, 15);
      acc[t] = __builtin_amdgcn_wmma_f32_16x16x32_bf16(
          false, ah, false, bh, (short)0, acc[t], false, false);
      acc[t] = __builtin_amdgcn_wmma_f32_16x16x32_bf16(
          false, ah, false, bl, (short)0, acc[t], false, false);
      acc[t] = __builtin_amdgcn_wmma_f32_16x16x32_bf16(
          false, al, false, bh, (short)0, acc[t], false, false);
    }
  }

  // ---- In-lane argmax over this wave's 128 codes (ascending order -> '>').
  float best = -3.402823466e38f;
  int   bidx = 0;
  #pragma unroll
  for (int t = 0; t < 8; ++t) {
    const int code0 = cbase + t * 16 + h2 * 8;
    #pragma unroll
    for (int r = 0; r < 8; ++r) {
      float s = acc[t][r];
      if (s > best) { best = s; bidx = code0 + r; }
    }
  }
  {  // merge lane halves (codes +0..7 vs +8..15) for token nrow
    float ob = __shfl_xor(best, 16);
    int   oi = __shfl_xor(bidx, 16);
    if (ob > best || (ob == best && oi < bidx)) { best = ob; bidx = oi; }
  }
  if (lane < 16) { sbest[g][nrow] = best; sidx[g][nrow] = bidx; }
  __syncthreads();

  // ---- Merge the 8 code groups (ascending ranges -> '>' keeps lowest idx).
  if (g == 0 && lane < 16) {
    float bb = sbest[0][nrow];
    int   bi = sidx[0][nrow];
    #pragma unroll
    for (int j = 1; j < 8; ++j) {
      float v  = sbest[j][nrow];
      int   vi = sidx[j][nrow];
      if (v > bb || (v == bb && vi < bi)) { bb = v; bi = vi; }
    }
    fidx[nrow] = bi;
    out[QUANT_N + 2 + m_base + nrow] = (float)bi;  // indices (flat f32 buffer)
  }
  __syncthreads();

  // ---- Cooperative exact-f32 gather + loss partial (16 rows x 128 float4).
  float sum = 0.f;
  #pragma unroll
  for (int it = 0; it < 8; ++it) {
    const int e   = it * 256 + tid;
    const int m   = e >> 7;
    const int c   = e & 127;
    const int idx = fidx[m];
    float4 wv = ((const float4*)(w + (size_t)idx * DIM))[c];
    float4 xv = ((const float4*)(x + (size_t)(m_base + m) * DIM))[c];
    ((float4*)(out + (size_t)(m_base + m) * DIM))[c] = wv;
    float d0 = wv.x - xv.x, d1 = wv.y - xv.y;
    float d2 = wv.z - xv.z, d3 = wv.w - xv.w;
    sum += d0 * d0 + d1 * d1 + d2 * d2 + d3 * d3;
  }
  #pragma unroll
  for (int off = 16; off > 0; off >>= 1) sum += __shfl_xor(sum, off);
  if (lane == 0) ssum[g] = sum;
  __syncthreads();
  if (tid == 0) {
    float bs = 0.f;
    #pragma unroll
    for (int j = 0; j < 8; ++j) bs += ssum[j];
    partial[blockIdx.x] = bs;   // deterministic two-stage reduction
  }
}

// ---------------------------------------------------------------------------
// Kernel 3: fold 1024 block partials -> mean; both losses are numerically
// identical (stop_gradient only changes gradients).
// ---------------------------------------------------------------------------
__global__ void vq_finalize(const float* __restrict__ partial,
                            float* __restrict__ out) {
  const int lane = threadIdx.x & 31;
  float s = 0.f;
  for (int i = lane; i < 1024; i += 32) s += partial[i];
  #pragma unroll
  for (int off = 16; off > 0; off >>= 1) s += __shfl_xor(s, off);
  if (lane == 0) {
    float mean = s / (float)QUANT_N;
    out[QUANT_N]     = mean;  // codebook_loss
    out[QUANT_N + 1] = mean;  // commitment_loss
  }
}

extern "C" void kernel_launch(void* const* d_in, const int* in_sizes, int n_in,
                              void* d_out, int out_size, void* d_ws, size_t ws_size,
                              hipStream_t stream) {
  const float* x = (const float*)d_in[0];
  const float* w = (const float*)d_in[1];
  float* out = (float*)d_out;

  // Workspace layout (needs ~2.01 MB):
  //   [0)      nh2      1024 f32
  //   [4KB)    partial  1024 f32
  //   [8KB)    w_hi     1024*512 bf16 (1 MB)
  //   [8KB+1M) w_lo     1024*512 bf16 (1 MB)
  char* ws = (char*)d_ws;
  float*  nh2     = (float*)ws;
  float*  partial = (float*)(ws + 4096);
  __bf16* w_hi    = (__bf16*)(ws + 8192);
  __bf16* w_lo    = (__bf16*)(ws + 8192 + (size_t)K_CODES * DIM * 2);

  vq_prep<<<K_CODES / 8, 256, 0, stream>>>(w, nh2, w_hi, w_lo);
  vq_main<<<M_TOTAL / 16, 256, 0, stream>>>(x, w, nh2, w_hi, w_lo, out, partial);
  vq_finalize<<<1, 32, 0, stream>>>(partial, out);
}
